// DGCNNEncoderGn_63410897158189
// MI455X (gfx1250) — matrix operand order repaired and checked by reading, not compiled, not run on hardware
//
#include <hip/hip_runtime.h>
#include <hip/hip_bf16.h>

typedef __attribute__((ext_vector_type(16))) _Float16 v16h;
typedef __attribute__((ext_vector_type(8)))  float    v8f;
typedef __attribute__((ext_vector_type(4)))  unsigned int u32x4;
typedef __attribute__((ext_vector_type(8)))  int      i32x8;
typedef __attribute__((ext_vector_type(4)))  int      i32x4;

#define BATCH 4
#define NPTS  2048
#define KNN   80
#define EPSF  1e-5f

// ---------------------------------------------------------------- utilities
__global__ void zero_f32_kernel(float* p, int n) {
  int i = blockIdx.x * blockDim.x + threadIdx.x;
  if (i < n) p[i] = 0.f;
}

__global__ void conv_f16_kernel(const float* src, _Float16* dst, int n) {
  int i = blockIdx.x * blockDim.x + threadIdx.x;
  if (i < n) dst[i] = (_Float16)src[i];
}

// Build Wcat = [Wl ; Wr - Wl] as f16, K padded with zeros to Kpad.
// W is [O, 2C] row-major. dst is [2O, Kpad].
__global__ void prep_wcat_kernel(const float* W, _Float16* dst, int O, int C, int Kpad) {
  int i = blockIdx.x * blockDim.x + threadIdx.x;
  int total = 2 * O * Kpad;
  if (i >= total) return;
  int r = i / Kpad, k = i % Kpad;
  float v = 0.f;
  if (k < C) {
    v = (r < O) ? W[r * (2 * C) + k]
                : (W[(r - O) * (2 * C) + C + k] - W[(r - O) * (2 * C) + k]);
  }
  dst[i] = (_Float16)v;
}

// x [B,3,N] channel-first -> xh [B,N,32] f16 (padded), sq[b][n] = |x|^2 (of f16-rounded vals)
__global__ void build_xh1_kernel(const float* x, _Float16* xh, float* sq, int N) {
  int b = blockIdx.y;
  int n = blockIdx.x * blockDim.x + threadIdx.x;
  if (n >= N) return;
  const float* xb = x + (long)b * 3 * N;
  _Float16 h[3];
  float s = 0.f;
  for (int c = 0; c < 3; ++c) {
    _Float16 hv = (_Float16)xb[c * N + n];
    h[c] = hv;
    float vf = (float)hv;
    s += vf * vf;
  }
  _Float16* row = xh + ((long)b * N + n) * 32;
  for (int k = 0; k < 32; ++k) row[k] = (k < 3) ? h[k] : (_Float16)0.f;
  sq[(long)b * N + n] = s;
}

// ------------------------------------------------- strided-batch WMMA GEMM
// out[b][m][n] = sum_k A[b*sAb + m*lda + k] * B[b*sBb + n*ldb + k] (+ bias[n])
// One wave per 16(M) x 64(N) strip: A fragment reused across 4 WMMAs per K-step.
// Kd % 32 == 0, M % 16 == 0, N % 64 == 0.
__global__ __launch_bounds__(128)
void wmma_gemm_f16(const _Float16* __restrict__ A, const _Float16* __restrict__ Bm,
                   float* __restrict__ C, const float* __restrict__ bias,
                   int M, int N, int Kd, int lda, int ldb, int ldc,
                   long sAb, long sBb, long sCb, int nBatch) {
  int wavesPerBlock = blockDim.x >> 5;
  int gwave = blockIdx.x * wavesPerBlock + (threadIdx.x >> 5);
  int lane  = threadIdx.x & 31;
  int stripsN = N >> 6, tilesM = M >> 4;
  int perBatch = tilesM * stripsN;
  int b = gwave / perBatch;
  if (b >= nBatch) return;
  int t  = gwave % perBatch;
  int tm = t / stripsN, ts = t % stripsN;

  const _Float16* Ab = A + (long)b * sAb;
  const _Float16* Bb = Bm + (long)b * sBb;
  float* Cb = C + (long)b * sCb;

  int grp   = lane >> 4;                 // K-half selector
  int mrow  = tm * 16 + (lane & 15);     // A row for this lane
  int ncol0 = ts * 64 + (lane & 15);     // first B column for this lane

  v8f acc[4] = {v8f{}, v8f{}, v8f{}, v8f{}};
  const _Float16* arow = Ab + (long)mrow * lda;
  for (int kk = 0; kk < Kd; kk += 32) {
    // A 16x32 f16 fragment: lanes 0-15 K=0..15, lanes 16-31 K=16..31 (matches B)
    v16h a;
#pragma unroll
    for (int j = 0; j < 8; ++j) {
      int k0 = kk + 2 * j + grp * 16;
      a[2 * j]     = arow[k0];
      a[2 * j + 1] = arow[k0 + 1];
    }
#pragma unroll
    for (int i = 0; i < 4; ++i) {
      // B 32x16 fragment: column per lane, 16 contiguous K values (32B aligned load)
      const _Float16* bp = Bb + (long)(ncol0 + i * 16) * ldb + kk + grp * 16;
      v16h bf = *(const v16h*)bp;
      acc[i] = __builtin_amdgcn_wmma_f32_16x16x32_f16(false, a, false, bf,
                                                      (short)0, acc[i], false, false);
    }
  }
  // C/D layout: element j -> row tm*16 + j + 8*grp, col = ncol
#pragma unroll
  for (int i = 0; i < 4; ++i) {
    int ccol = ncol0 + i * 16;
    float bv = bias ? bias[ccol] : 0.f;
#pragma unroll
    for (int j = 0; j < 8; ++j) {
      int crow = tm * 16 + j + 8 * grp;
      Cb[(long)crow * ldc + ccol] = acc[i][j] + bv;
    }
  }
}

// ------------------------------------------------------------- top-K (LDS bitonic)
// neg_dist[i][j] = 2*gram[i][j] - sq[i] - sq[j]; sort row descending, keep 80 indices.
// The 8KB gram row is staged into LDS by the Tensor Data Mover (wave 0 issues one
// 1-D descriptor, waits TENSORcnt, then the workgroup barrier publishes the data).
__global__ __launch_bounds__(1024)
void topk_kernel(const float* __restrict__ gram, const float* __restrict__ sq,
                 int* __restrict__ idxout, int N, int Ksel) {
  __shared__ float vals[NPTS];
  __shared__ int   inds[NPTS];
  int b = blockIdx.y, row = blockIdx.x;
  const float* g   = gram + ((long)b * N + row) * N;
  const float* sqb = sq + (long)b * N;
  float sqr = sqb[row];

#if __has_builtin(__builtin_amdgcn_tensor_load_to_lds)
  if ((threadIdx.x >> 5) == 0) {   // wave 0 only: one TDM op per row
    unsigned ldsOff = (unsigned)(uintptr_t)(void*)vals;          // LDS byte offset
    unsigned long long ga = (unsigned long long)(uintptr_t)g;    // global byte addr
    u32x4 g0;
    g0[0] = 1u;                                                  // count=1 (valid, user mode)
    g0[1] = ldsOff;                                              // lds_addr
    g0[2] = (unsigned)(ga & 0xffffffffu);                        // global_addr[31:0]
    g0[3] = (unsigned)((ga >> 32) & 0x1ffffffu) | (2u << 30);    // addr[56:32] | type=2
    i32x8 g1;
    g1[0] = (int)(2u << 16);                       // data_size = 4B
    g1[1] = (int)((unsigned)(NPTS & 0xffff) << 16);// tensor_dim0 = 2048 (bits 79:48)
    g1[2] = (int)(1u << 16);                       // tensor_dim0 hi=0; tensor_dim1 = 1
    g1[3] = (int)((unsigned)(NPTS & 0xffff) << 16);// tile_dim0 = 2048 (bits 127:112)
    g1[4] = 0;                                     // tile_dim1 = tile_dim2 = 0 (unused)
    g1[5] = NPTS;                                  // tensor_dim0_stride
    g1[6] = 0; g1[7] = 0;
    i32x4 gz = {0, 0, 0, 0};
#if defined(__clang_major__) && (__clang_major__ >= 23)
    i32x8 gz8 = {0, 0, 0, 0, 0, 0, 0, 0};
    __builtin_amdgcn_tensor_load_to_lds(g0, g1, gz, gz, gz8, 0);
#else
    __builtin_amdgcn_tensor_load_to_lds(g0, g1, gz, gz, 0);
#endif
    __builtin_amdgcn_s_wait_tensorcnt(0);
  }
  __syncthreads();
  for (int j = threadIdx.x; j < N; j += blockDim.x) {
    vals[j] = 2.f * vals[j] - sqr - sqb[j];
    inds[j] = j;
  }
#else
  for (int j = threadIdx.x; j < N; j += blockDim.x) {
    vals[j] = 2.f * g[j] - sqr - sqb[j];
    inds[j] = j;
  }
#endif
  __syncthreads();
  for (int k = 2; k <= N; k <<= 1) {
    for (int jj = k >> 1; jj > 0; jj >>= 1) {
      for (int t = threadIdx.x; t < N; t += blockDim.x) {
        int ixj = t ^ jj;
        if (ixj > t) {
          float va = vals[t], vb = vals[ixj];
          int   ia = inds[t], ib = inds[ixj];
          bool aFirst = (va > vb) || (va == vb && ia < ib);  // descending, ties -> low idx
          bool desc = ((t & k) == 0);
          if (desc ? !aFirst : aFirst) {
            vals[t] = vb; vals[ixj] = va;
            inds[t] = ib; inds[ixj] = ia;
          }
        }
      }
      __syncthreads();
    }
  }
  for (int t = threadIdx.x; t < Ksel; t += blockDim.x)
    idxout[((long)b * N + row) * Ksel + t] = inds[t];
}

// ---------------------------------------------- EdgeConv gather-reduce + GN stats
// yz [B,N,2O] (y | z). h[b,n,k,o] = y[idx[k]][o] + z[n][o].
// Track max_k and min_k, accumulate sum/sumsq per (b, group) with float atomics.
__global__ void edge_reduce_kernel(const float* __restrict__ yz, const int* __restrict__ idx,
                                   float* __restrict__ hmm, float* __restrict__ stats,
                                   int N, int Ksel, int O, int groups) {
  __shared__ int   sidx[KNN];
  __shared__ float red[128];
  int b = blockIdx.y, n = blockIdx.x, o = threadIdx.x;
  const float* yb = yz + (long)b * N * 2 * O;
  const int*   ip = idx + ((long)b * N + n) * Ksel;
  for (int t = o; t < Ksel; t += blockDim.x) sidx[t] = ip[t];
  __syncthreads();
  float z = yb[(long)n * 2 * O + O + o];
  float hmax = -__builtin_inff(), hmin = __builtin_inff(), s = 0.f, s2 = 0.f;
  for (int k = 0; k < Ksel; ++k) {
    if (k + 1 < Ksel)
      __builtin_prefetch(&yb[(long)sidx[k + 1] * 2 * O + o], 0, 1);  // global_prefetch_b8
    float v = yb[(long)sidx[k] * 2 * O + o] + z;
    hmax = fmaxf(hmax, v); hmin = fminf(hmin, v);
    s += v; s2 += v * v;
  }
  float* hrow = hmm + ((long)b * N + n) * 2 * O;
  hrow[o] = hmax; hrow[O + o] = hmin;
  int Cg = O / groups, g = o / Cg, lo = o % Cg;
  red[o] = s; __syncthreads();
  for (int st = Cg >> 1; st > 0; st >>= 1) { if (lo < st) red[o] += red[o + st]; __syncthreads(); }
  if (lo == 0) atomicAdd(&stats[(b * groups + g) * 2 + 0], red[o]);
  __syncthreads();
  red[o] = s2; __syncthreads();
  for (int st = Cg >> 1; st > 0; st >>= 1) { if (lo < st) red[o] += red[o + st]; __syncthreads(); }
  if (lo == 0) atomicAdd(&stats[(b * groups + g) * 2 + 1], red[o]);
}

// Apply GN + LeakyReLU + (max via monotonicity); emit f32 + f16 feats and |x|^2 for next kNN.
__global__ void edge_apply_kernel(const float* __restrict__ hmm, const float* __restrict__ stats,
                                  const float* __restrict__ gamma, const float* __restrict__ beta,
                                  float* __restrict__ feats, _Float16* __restrict__ featsh,
                                  float* __restrict__ sqout,
                                  int N, int Ksel, int O, int groups, int fstride, int foff) {
  __shared__ float red[128];
  int b = blockIdx.y, n = blockIdx.x, o = threadIdx.x;
  int Cg = O / groups, g = o / Cg;
  float cnt  = (float)N * (float)Ksel * (float)Cg;
  float mean = stats[(b * groups + g) * 2 + 0] / cnt;
  float var  = stats[(b * groups + g) * 2 + 1] / cnt - mean * mean;
  float inv  = rsqrtf(var + EPSF);
  float scale = gamma[o] * inv;
  float shift = beta[o] - mean * scale;
  const float* hrow = hmm + ((long)b * N + n) * 2 * O;
  float sel = (scale >= 0.f) ? hrow[o] : hrow[O + o];
  float v = scale * sel + shift;
  v = (v >= 0.f) ? v : 0.2f * v;                 // LeakyReLU(0.2)
  long fo = ((long)b * N + n) * fstride + foff + o;
  feats[fo] = v;
  _Float16 hv = (_Float16)v;
  featsh[fo] = hv;
  float vf = (float)hv;
  red[o] = vf * vf; __syncthreads();
  for (int st = blockDim.x >> 1; st > 0; st >>= 1) { if (o < st) red[o] += red[o + st]; __syncthreads(); }
  if (o == 0 && sqout) sqout[(long)b * N + n] = red[0];
}

// -------------------------------------------------------- final GN stats / reduce
__global__ void final_stats_kernel(const float* __restrict__ h, float* __restrict__ mstats,
                                   int N, int O, int groups) {
  __shared__ float sred[8], s2red[8];
  int b = blockIdx.y, n = blockIdx.x;
  if (threadIdx.x < 8) { sred[threadIdx.x] = 0.f; s2red[threadIdx.x] = 0.f; }
  __syncthreads();
  const float* hr = h + ((long)b * N + n) * O;
  int per = O / blockDim.x;                      // 4 channels/thread, same group
  int o0 = threadIdx.x * per;
  int g = o0 / (O / groups);
  float s = 0.f, s2 = 0.f;
  for (int j = 0; j < per; ++j) { float v = hr[o0 + j]; s += v; s2 += v * v; }
  atomicAdd(&sred[g], s); atomicAdd(&s2red[g], s2);
  __syncthreads();
  if (threadIdx.x < 8) {
    atomicAdd(&mstats[(b * groups + threadIdx.x) * 2 + 0], sred[threadIdx.x]);
    atomicAdd(&mstats[(b * groups + threadIdx.x) * 2 + 1], s2red[threadIdx.x]);
  }
}

__global__ void final_reduce_kernel(const float* __restrict__ h, const float* __restrict__ mstats,
                                    const float* __restrict__ gm, const float* __restrict__ bg,
                                    float* __restrict__ x4, int N, int O, int groups) {
  int b = blockIdx.y;
  int o = blockIdx.x * blockDim.x + threadIdx.x;
  float hmax = -__builtin_inff(), hmin = __builtin_inff();
  for (int n = 0; n < N; ++n) {
    float v = h[((long)b * N + n) * O + o];      // coalesced across threads
    hmax = fmaxf(hmax, v); hmin = fminf(hmin, v);
  }
  int Cg = O / groups, g = o / Cg;
  float cnt  = (float)N * (float)Cg;
  float mean = mstats[(b * groups + g) * 2 + 0] / cnt;
  float var  = mstats[(b * groups + g) * 2 + 1] / cnt - mean * mean;
  float inv  = rsqrtf(var + EPSF);
  float scale = gm[o] * inv;
  float shift = bg[o] - mean * scale;
  float sel = (scale >= 0.f) ? hmax : hmin;
  x4[(long)b * O + o] = fmaxf(scale * sel + shift, 0.f);   // ReLU then max == max then ReLU
}

// ------------------------------------------------------------------ transpose
__global__ void transpose_feats_kernel(const float* __restrict__ feats, float* __restrict__ out,
                                       int N, int C) {
  __shared__ float tile[32][33];
  int b = blockIdx.z;
  int n0 = blockIdx.x * 32, c0 = blockIdx.y * 32;
  for (int i = threadIdx.y; i < 32; i += blockDim.y)
    tile[i][threadIdx.x] = feats[((long)b * N + n0 + i) * C + c0 + threadIdx.x];
  __syncthreads();
  for (int i = threadIdx.y; i < 32; i += blockDim.y)
    out[((long)b * C + c0 + i) * N + n0 + threadIdx.x] = tile[threadIdx.x][i];
}

// ------------------------------------------------------------------ launcher
static inline size_t alignUp256(size_t x) { return (x + 255) & ~(size_t)255; }

extern "C" void kernel_launch(void* const* d_in, const int* in_sizes, int n_in,
                              void* d_out, int out_size, void* d_ws, size_t ws_size,
                              hipStream_t stream) {
  const float* x  = (const float*)d_in[0];
  const float* W1 = (const float*)d_in[1];
  const float* g1 = (const float*)d_in[2];
  const float* b1 = (const float*)d_in[3];
  const float* W2 = (const float*)d_in[4];
  const float* g2 = (const float*)d_in[5];
  const float* b2 = (const float*)d_in[6];
  const float* W3 = (const float*)d_in[7];
  const float* g3 = (const float*)d_in[8];
  const float* b3 = (const float*)d_in[9];
  const float* Wm = (const float*)d_in[10];
  const float* bm = (const float*)d_in[11];
  const float* gm = (const float*)d_in[12];
  const float* bg = (const float*)d_in[13];
  float* out = (float*)d_out;

  const int B = BATCH, N = NPTS, Ks = KNN;

  // workspace carve-out (256B aligned)
  char* ws = (char*)d_ws;
  size_t off = 0;
  auto take = [&](size_t bytes) { char* p = ws + off; off = alignUp256(off + bytes); return p; };
  _Float16* xh1    = (_Float16*)take((size_t)B * N * 32 * 2);
  _Float16* featsh = (_Float16*)take((size_t)B * N * 256 * 2);
  float*    feats  = (float*)   take((size_t)B * N * 256 * 4);
  float*    gram   = (float*)   take((size_t)B * N * N * 4);
  float*    sq     = (float*)   take((size_t)B * N * 4);
  int*      knnIdx = (int*)     take((size_t)B * N * Ks * 4);
  float*    yz     = (float*)   take((size_t)B * N * 256 * 4);
  float*    hmm    = (float*)   take((size_t)B * N * 256 * 4);
  float*    hfin   = (float*)   take((size_t)B * N * 1024 * 4);
  _Float16* wc1    = (_Float16*)take((size_t)128 * 32 * 2);
  _Float16* wc2    = (_Float16*)take((size_t)128 * 64 * 2);
  _Float16* wc3    = (_Float16*)take((size_t)256 * 64 * 2);
  _Float16* wmh    = (_Float16*)take((size_t)1024 * 256 * 2);
  float*    st1    = (float*)   take(1024 * 4);   // st1 | st2 | st3 | mstats all inside
  float*    st2    = st1 + 64;
  float*    st3    = st1 + 128;
  float*    mst    = st1 + 256;

  // zero all GN stat accumulators
  zero_f32_kernel<<<4, 256, 0, stream>>>(st1, 1024);

  // weight prep
  prep_wcat_kernel<<<(2 * 64 * 32 + 255) / 256, 256, 0, stream>>>(W1, wc1, 64, 3, 32);
  prep_wcat_kernel<<<(2 * 64 * 64 + 255) / 256, 256, 0, stream>>>(W2, wc2, 64, 64, 64);
  prep_wcat_kernel<<<(2 * 128 * 64 + 255) / 256, 256, 0, stream>>>(W3, wc3, 128, 64, 64);
  conv_f16_kernel<<<(1024 * 256 + 255) / 256, 256, 0, stream>>>(Wm, wmh, 1024 * 256);

  // x [B,3,N] -> xh1 [B,N,32] f16 + sq
  build_xh1_kernel<<<dim3(N / 256, B), 256, 0, stream>>>(x, xh1, sq, N);

  auto gemm = [&](const _Float16* A, const _Float16* Bm2, float* C, const float* bias,
                  int M, int Nn, int Kd, int lda, int ldb, int ldc,
                  long sAb, long sBb, long sCb) {
    int waves = B * (M / 16) * (Nn / 64);   // one wave per 16x64 strip
    wmma_gemm_f16<<<waves / 4, 128, 0, stream>>>(A, Bm2, C, bias, M, Nn, Kd,
                                                 lda, ldb, ldc, sAb, sBb, sCb, B);
  };

  // ---------------- layer 1 (C=3 padded to 32, O=64, groups=2) ----------------
  gemm(xh1, xh1, gram, nullptr, N, N, 32, 32, 32, N, (long)N * 32, (long)N * 32, (long)N * N);
  topk_kernel<<<dim3(N, B), 1024, 0, stream>>>(gram, sq, knnIdx, N, Ks);
  gemm(xh1, wc1, yz, nullptr, N, 128, 32, 32, 32, 128, (long)N * 32, 0, (long)N * 128);
  edge_reduce_kernel<<<dim3(N, B), 64, 0, stream>>>(yz, knnIdx, hmm, st1, N, Ks, 64, 2);
  edge_apply_kernel<<<dim3(N, B), 64, 0, stream>>>(hmm, st1, g1, b1, feats, featsh, sq,
                                                   N, Ks, 64, 2, 256, 0);

  // ---------------- layer 2 (C=64, O=64, groups=2) ----------------
  gemm(featsh, featsh, gram, nullptr, N, N, 64, 256, 256, N,
       (long)N * 256, (long)N * 256, (long)N * N);
  topk_kernel<<<dim3(N, B), 1024, 0, stream>>>(gram, sq, knnIdx, N, Ks);
  gemm(featsh, wc2, yz, nullptr, N, 128, 64, 256, 64, 128, (long)N * 256, 0, (long)N * 128);
  edge_reduce_kernel<<<dim3(N, B), 64, 0, stream>>>(yz, knnIdx, hmm, st2, N, Ks, 64, 2);
  edge_apply_kernel<<<dim3(N, B), 64, 0, stream>>>(hmm, st2, g2, b2, feats, featsh, sq,
                                                   N, Ks, 64, 2, 256, 64);

  // ---------------- layer 3 (C=64, O=128, groups=4) ----------------
  gemm(featsh + 64, featsh + 64, gram, nullptr, N, N, 64, 256, 256, N,
       (long)N * 256, (long)N * 256, (long)N * N);
  topk_kernel<<<dim3(N, B), 1024, 0, stream>>>(gram, sq, knnIdx, N, Ks);
  gemm(featsh + 64, wc3, yz, nullptr, N, 256, 64, 256, 64, 256, (long)N * 256, 0, (long)N * 256);
  edge_reduce_kernel<<<dim3(N, B), 128, 0, stream>>>(yz, knnIdx, hmm, st3, N, Ks, 128, 4);
  edge_apply_kernel<<<dim3(N, B), 128, 0, stream>>>(hmm, st3, g3, b3, feats, featsh, nullptr,
                                                    N, Ks, 128, 4, 256, 128);

  // ---------------- final projection + GN(8) + ReLU + max over N ----------------
  gemm(featsh, wmh, hfin, bm, N, 1024, 256, 256, 256, 1024, (long)N * 256, 0, (long)N * 1024);
  final_stats_kernel<<<dim3(N, B), 256, 0, stream>>>(hfin, mst, N, 1024, 8);
  final_reduce_kernel<<<dim3(4, B), 256, 0, stream>>>(hfin, mst, gm, bg, out, N, 1024, 8);

  // x_features = transpose(feats) -> out + 4096
  transpose_feats_kernel<<<dim3(N / 32, 256 / 32, B), dim3(32, 8), 0, stream>>>(
      feats, out + (size_t)B * 1024, N, 256);
}
